// GraphSAGE_29695403885028
// MI455X (gfx1250) — matrix operand order repaired and checked by reading, not compiled, not run on hardware
//
#include <hip/hip_runtime.h>

#define G    512
#define NPG  90
#define NN   (G*NPG)       // 46080
#define EDGES (NN*20)      // 921600
#define F_IN 90
#define HC   64
#define KPOOL 70

typedef __attribute__((ext_vector_type(16))) _Float16 v16h;
typedef __attribute__((ext_vector_type(8)))  float    v8f;

// ---------------- zero ----------------
__global__ void zero_f32(float* __restrict__ p, int n) {
    int i = blockIdx.x * blockDim.x + threadIdx.x;
    if (i < n) p[i] = 0.0f;
}

// ---------------- dual projection via WMMA f16 ----------------
// P[N,64] = A @ Wl ; S[N,64] = A @ Wr   (shared A fragments, 2x reuse)
// 8 waves/block, 1 wave = one 16-row M tile x 64 cols x 2 outputs.
template<int KF>
__global__ __launch_bounds__(256) void proj_wmma(
    const float* __restrict__ Asrc,
    const float* __restrict__ Wl, const float* __restrict__ Wr,
    float* __restrict__ P, float* __restrict__ S) {
    const int lane = threadIdx.x & 31;
    const int wave = threadIdx.x >> 5;
    const int row0 = (blockIdx.x * 8 + wave) * 16;
    if (row0 >= NN) return;

    const int rA     = row0 + (lane & 15);
    const int kbaseA = (lane < 16) ? 0 : 8;
    const int colB   = lane & 15;
    const int koffB  = (lane < 16) ? 0 : 8;
    const int NKB    = (KF + 31) / 32;

    v8f cL[4] = {};
    v8f cR[4] = {};

    for (int kb = 0; kb < NKB; ++kb) {
        v16h a;
        #pragma unroll
        for (int i = 0; i < 8; ++i) {
            int k0 = kb * 32 + kbaseA + i;
            int k1 = k0 + 16;
            a[i]     = (k0 < KF) ? (_Float16)Asrc[(long)rA * KF + k0] : (_Float16)0.0f;
            a[i + 8] = (k1 < KF) ? (_Float16)Asrc[(long)rA * KF + k1] : (_Float16)0.0f;
        }
        #pragma unroll
        for (int nt = 0; nt < 4; ++nt) {
            v16h bL, bR;
            #pragma unroll
            for (int i = 0; i < 16; ++i) {
                int k = kb * 32 + koffB + (i & 7) + ((i >> 3) << 4);
                bL[i] = (k < KF) ? (_Float16)Wl[(long)k * HC + nt * 16 + colB] : (_Float16)0.0f;
                bR[i] = (k < KF) ? (_Float16)Wr[(long)k * HC + nt * 16 + colB] : (_Float16)0.0f;
            }
            cL[nt] = __builtin_amdgcn_wmma_f32_16x16x32_f16(
                         false, a, false, bL, (short)0, cL[nt], false, false);
            cR[nt] = __builtin_amdgcn_wmma_f32_16x16x32_f16(
                         false, a, false, bR, (short)0, cR[nt], false, false);
        }
    }

    const int rowAdd = (lane < 16) ? 0 : 8;
    #pragma unroll
    for (int nt = 0; nt < 4; ++nt) {
        #pragma unroll
        for (int v = 0; v < 8; ++v) {
            long idx = (long)(row0 + v + rowAdd) * HC + nt * 16 + colB;
            P[idx] = cL[nt][v];
            S[idx] = cR[nt][v];
        }
    }
}

// ---------------- edge scatter of projected rows (64-wide) ----------------
// acc[dst] += P[src]; optionally count degree. Accumulators are L2-resident.
template<bool COUNT_DEG>
__global__ void scatter64(const float* __restrict__ P, const int* __restrict__ ei,
                          float* __restrict__ deg, float* __restrict__ acc) {
    int e = blockIdx.x * blockDim.x + threadIdx.x;
    if (e >= EDGES) return;
    int src = ei[e];
    int dst = ei[EDGES + e];
    if (COUNT_DEG) atomicAdd(&deg[dst], 1.0f);
    const float* ps = P + (long)src * HC;
    float* ad = acc + (long)dst * HC;
    #pragma unroll 8
    for (int f = 0; f < HC; ++f) atomicAdd(&ad[f], ps[f]);
}

// ---------------- combine: h = acc*rdeg + S + bias (+relu copy) -----------
template<bool WRITE_RELU>
__global__ void combine64(const float* __restrict__ acc, const float* __restrict__ S,
                          const float* __restrict__ deg, const float* __restrict__ bias,
                          float* __restrict__ outp, float* __restrict__ relup) {
    int i = blockIdx.x * blockDim.x + threadIdx.x;
    if (i >= NN * HC) return;
    int row = i >> 6;
    int col = i & 63;
    float rdeg = 1.0f / fmaxf(deg[row], 1.0f);
    float val = acc[i] * rdeg + S[i] + bias[col];
    outp[i] = val;
    if (WRITE_RELU) relup[i] = fmaxf(val, 0.0f);
}

// ---------------- sort-pool ranking (stable argsort of -x, top 70) ----------
__global__ __launch_bounds__(96) void sortpool_rank(const float* __restrict__ out2,
                                                    int* __restrict__ sel) {
    __shared__ float v[NPG];
    const int g = blockIdx.x;
    const int t = threadIdx.x;
    if (t < NPG) v[t] = out2[((long)g * NPG + t) * HC + (HC - 1)];
    __syncthreads();
    if (t < NPG) {
        float mv = v[t];
        int rank = 0;
        for (int j = 0; j < NPG; ++j) {
            float o = v[j];
            rank += (o > mv) || (o == mv && j < t);
        }
        if (rank < KPOOL) sel[g * KPOOL + rank] = t;
    }
}

// ---------------- pooled MLP: 16 graphs/wave, K=4480, fused head ------------
__global__ __launch_bounds__(32) void mlp_wmma(
    const float* __restrict__ out2, const int* __restrict__ sel,
    const float* __restrict__ Wl1, const float* __restrict__ bl1,
    const float* __restrict__ Wl2, const float* __restrict__ bl2,
    float* __restrict__ outp) {
    __shared__ int   selS[16 * KPOOL];
    __shared__ float Ps[16 * HC];
    const int lane = threadIdx.x;
    const int g0   = blockIdx.x * 16;

    for (int i = lane; i < 16 * KPOOL; i += 32) {
        int gl = i / KPOOL, r = i % KPOOL;
        selS[i] = sel[(g0 + gl) * KPOOL + r];
    }
    __syncthreads();

    const int gl     = lane & 15;
    const int kbaseA = (lane < 16) ? 0 : 8;
    const int colB   = lane & 15;
    const int koffB  = (lane < 16) ? 0 : 8;
    const long gbase = (long)(g0 + gl) * NPG;

    v8f c[4] = {};
    for (int kb = 0; kb < (KPOOL * HC) / 32; ++kb) {   // 140 k-steps
        v16h a;
        #pragma unroll
        for (int i = 0; i < 8; ++i) {
            int k0 = kb * 32 + kbaseA + i;
            int k1 = k0 + 16;
            int r0 = k0 >> 6, c0 = k0 & 63;
            int r1 = k1 >> 6, c1 = k1 & 63;
            a[i]     = (_Float16)out2[(gbase + selS[gl * KPOOL + r0]) * HC + c0];
            a[i + 8] = (_Float16)out2[(gbase + selS[gl * KPOOL + r1]) * HC + c1];
        }
        #pragma unroll
        for (int nt = 0; nt < 4; ++nt) {
            v16h b;
            #pragma unroll
            for (int i = 0; i < 16; ++i) {
                int k = kb * 32 + koffB + (i & 7) + ((i >> 3) << 4);
                b[i] = (_Float16)Wl1[(long)k * HC + nt * 16 + colB];
            }
            c[nt] = __builtin_amdgcn_wmma_f32_16x16x32_f16(
                        false, a, false, b, (short)0, c[nt], false, false);
        }
    }

    const int rowAdd = (lane < 16) ? 0 : 8;
    #pragma unroll
    for (int nt = 0; nt < 4; ++nt) {
        #pragma unroll
        for (int v = 0; v < 8; ++v) {
            int row = v + rowAdd;
            int col = nt * 16 + colB;
            Ps[row * HC + col] = c[nt][v] + bl1[col];
        }
    }
    __syncthreads();
    if (lane < 16) {
        float acc = 0.0f;
        #pragma unroll 8
        for (int ch = 0; ch < HC; ++ch) acc += Ps[lane * HC + ch] * Wl2[ch];
        acc += bl2[0];
        outp[g0 + lane] = 1.0f / (1.0f + __expf(-acc));
    }
}

extern "C" void kernel_launch(void* const* d_in, const int* in_sizes, int n_in,
                              void* d_out, int out_size, void* d_ws, size_t ws_size,
                              hipStream_t stream) {
    (void)in_sizes; (void)n_in; (void)out_size; (void)ws_size;
    const float* x   = (const float*)d_in[0];
    const int*   ei  = (const int*)  d_in[1];   // [2, E] (src row, dst row)
    // d_in[2] = batch (implied by contiguous layout; unused)
    const float* W1l = (const float*)d_in[3];
    const float* W1r = (const float*)d_in[4];
    const float* b1  = (const float*)d_in[5];
    const float* W2l = (const float*)d_in[6];
    const float* W2r = (const float*)d_in[7];
    const float* b2  = (const float*)d_in[8];
    const float* Wl1 = (const float*)d_in[9];
    const float* bl1 = (const float*)d_in[10];
    const float* Wl2 = (const float*)d_in[11];
    const float* bl2 = (const float*)d_in[12];

    // workspace (buffers reused between conv1 and conv2):
    float* ws    = (float*)d_ws;
    float* deg   = ws;                           // [N]
    float* acc   = deg   + NN;                   // [N*64] sum1 then sum2 (L2-resident)
    float* P     = acc   + (long)NN * HC;        // [N*64] projected-neighbor term
    float* S     = P     + (long)NN * HC;        // [N*64] self term
    float* relu1 = S     + (long)NN * HC;        // [N*64]
    float* out2  = relu1 + (long)NN * HC;        // [N*64]
    int*   sel   = (int*)(out2 + (long)NN * HC); // [G*70]

    float* out    = (float*)d_out;               // [512]
    float* xtrain = out + G;                     // [N*64] = conv1 pre-relu

    const int nElem = NN * HC;
    const int eBlocks = (EDGES + 255) / 256;
    const int gemmBlocks = NN / (16 * 8);        // 360

    // ---- conv1: project first (mean is linear), then 64-wide scatter ----
    zero_f32<<<(NN + nElem + 255) / 256, 256, 0, stream>>>(deg, NN + nElem);
    proj_wmma<F_IN><<<gemmBlocks, 256, 0, stream>>>(x, W1l, W1r, P, S);
    scatter64<true><<<eBlocks, 256, 0, stream>>>(P, ei, deg, acc);
    combine64<true><<<(nElem + 255) / 256, 256, 0, stream>>>(
        acc, S, deg, b1, xtrain, relu1);

    // ---- conv2: same scheme on relu(h) ----
    zero_f32<<<(nElem + 255) / 256, 256, 0, stream>>>(acc, nElem);
    proj_wmma<HC><<<gemmBlocks, 256, 0, stream>>>(relu1, W2l, W2r, P, S);
    scatter64<false><<<eBlocks, 256, 0, stream>>>(P, ei, deg, acc);
    combine64<false><<<(nElem + 255) / 256, 256, 0, stream>>>(
        acc, S, deg, b2, out2, nullptr);

    // ---- sort-pool top-70 selection per graph ----
    sortpool_rank<<<G, 96, 0, stream>>>(out2, sel);

    // ---- pooled MLP (16-graph x 64 GEMM, K=4480) fused with head + sigmoid ----
    mlp_wmma<<<G / 16, 32, 0, stream>>>(out2, sel, Wl1, bl1, Wl2, bl2, out);
}
